// TGAT_4561255268354
// MI455X (gfx1250) — compile-verified, matrix operand order checked by reading, not run on hardware
//
#include <hip/hip_runtime.h>
#include <hip/hip_bf16.h>
#include <math.h>

// ---------------- problem constants ----------------
#define N_NODESC 2048
#define SEQC     32
#define N_EDGESC 32768
#define EAC      (N_EDGESC + N_NODESC)   // edges + self loops = 34816
#define IN_CHC   128
#define HIDC     64
#define HEADSC   8
#define HC1C     (HEADSC * HIDC)         // 512
#define LSTM_HC  256
#define LSTM_GC  (4 * LSTM_HC)           // 1024
#define N_LAYERSC 4
#define N_CLASSESC 10
#define NEG_SLOPEC 0.2f

typedef __attribute__((ext_vector_type(16))) _Float16 v16h;
typedef __attribute__((ext_vector_type(8)))  float    v8f;

// ---------------- WMMA GEMM: C[M,N] = A[M,K] @ B[K,N] (+bias) (+=C) ----------------
// fp32 in/out, f16 WMMA compute, f32 accumulate (v_wmma_f32_16x16x32_f16).
// Block: 256 threads = 8 wave32s. Tile 128x64, BK=32, double-buffered LDS pipeline.
// Wave w: rows (w&3)*32, cols (w>>2)*32 -> 2x2 WMMA accumulators (32x32 per wave).
#define BM 128
#define BN 64
#define BK 32
#define LDS_S 36   // padded LDS stride in halves (even; pairs 4B aligned, quads 8B aligned)

template<bool FULL>
__global__ __launch_bounds__(256)
void gemm_wmma_f16(const float* __restrict__ A, const float* __restrict__ B,
                   const float* __restrict__ bias, float* __restrict__ C,
                   int M, int N, int K, int accumFlag, int biasFlag) {
    const int tid  = threadIdx.x;
    const int lane = tid & 31;
    const int wave = tid >> 5;
    const int bm = blockIdx.y * BM;
    const int bn = blockIdx.x * BN;
    const int wm = (wave & 3) * 32;
    const int wn = (wave >> 2) * 32;
    const int halfsel = lane >> 4;
    const int l15 = lane & 15;

    v8f acc00 = {}, acc01 = {}, acc10 = {}, acc11 = {};
    union Frag { v16h v; float f[8]; };

    if (FULL) {
        // =========== full-tile pipelined path (M%128==0, N%64==0, K%32==0) ===========
        __shared__ _Float16 As[2][BM * LDS_S];   // As[buf][m][k]
        __shared__ _Float16 Bs[2][BN * LDS_S];   // Bs[buf][n][k]

        // per-thread staging coordinates
        const int ar0 = tid >> 3;                // A rows for i=0..3: ar0 + i*32
        const int ac4 = (tid & 7) * 4;           // A k-offset
        const int bnn = tid & 63;                // B column within tile
        const int bkg = (tid >> 6) * 8;          // B k-group base (8 k's)

        float4 av[4];
        float  bv[8];

        // ---- fetch tile k0 into registers (loads only; no consumption) ----
        auto fetch = [&](int k0) {
            const float* Abase = A + (size_t)(bm + ar0) * K + k0 + ac4;
            #pragma unroll
            for (int i = 0; i < 4; ++i)
                av[i] = *(const float4*)(Abase + (size_t)i * 32 * K);
            const float* Bbase = B + (size_t)(k0 + bkg) * N + bn;   // uniform + lane n
            #pragma unroll
            for (int j = 0; j < 8; ++j)
                bv[j] = Bbase[(size_t)j * N + bnn];
        };
        // ---- convert + store registers into LDS buffer `buf` ----
        auto stage = [&](int buf) {
            #pragma unroll
            for (int i = 0; i < 4; ++i) {
                union { _Float16 h[4]; uint2 u; } p;
                p.h[0] = (_Float16)av[i].x; p.h[1] = (_Float16)av[i].y;
                p.h[2] = (_Float16)av[i].z; p.h[3] = (_Float16)av[i].w;
                *(uint2*)&As[buf][(ar0 + i * 32) * LDS_S + ac4] = p.u;   // 8B aligned
            }
            union { _Float16 h[8]; uint2 u[2]; } q;
            #pragma unroll
            for (int j = 0; j < 8; ++j) q.h[j] = (_Float16)bv[j];
            *(uint2*)&Bs[buf][bnn * LDS_S + bkg]     = q.u[0];
            *(uint2*)&Bs[buf][bnn * LDS_S + bkg + 4] = q.u[1];
        };

        fetch(0);
        stage(0);
        __syncthreads();

        int cur = 0;
        for (int k0 = 0; k0 < K; k0 += BK) {
            const bool hasNext = (k0 + BK) < K;
            if (hasNext) fetch(k0 + BK);                 // global loads in flight
            if (k0 + 2 * BK < K) {                       // prefetch tile after next
                int pr = bm + (tid >> 1);
                __builtin_prefetch(&A[(size_t)pr * K + (k0 + 2 * BK)], 0, 1);
                if (tid < 32)
                    __builtin_prefetch(&B[(size_t)(k0 + 2 * BK + tid) * N + bn], 0, 1);
            }

            // ---- fragments from current buffer; 2 A-frags x 2 B-frags -> 4 WMMA ----
            Frag a0, a1, b0, b1;
            #pragma unroll
            for (int q = 0; q < 8; ++q) {
                int kk = ((q & 4) ? 16 : 0) + halfsel * 8 + (q & 3) * 2;
                a0.f[q] = *(const float*)&As[cur][(wm + l15) * LDS_S + kk];
                a1.f[q] = *(const float*)&As[cur][(wm + 16 + l15) * LDS_S + kk];
                b0.f[q] = *(const float*)&Bs[cur][(wn + l15) * LDS_S + kk];
                b1.f[q] = *(const float*)&Bs[cur][(wn + 16 + l15) * LDS_S + kk];
            }
            acc00 = __builtin_amdgcn_wmma_f32_16x16x32_f16(false, a0.v, false, b0.v, (short)0, acc00, false, false);
            acc01 = __builtin_amdgcn_wmma_f32_16x16x32_f16(false, a0.v, false, b1.v, (short)0, acc01, false, false);
            acc10 = __builtin_amdgcn_wmma_f32_16x16x32_f16(false, a1.v, false, b0.v, (short)0, acc10, false, false);
            acc11 = __builtin_amdgcn_wmma_f32_16x16x32_f16(false, a1.v, false, b1.v, (short)0, acc11, false, false);

            if (hasNext) stage(cur ^ 1);                 // consume loads after WMMA issue
            __syncthreads();
            cur ^= 1;
        }
    } else {
        // =========== bounds-safe path (final FC only): clamped unconditional loads ===========
        __shared__ _Float16 As1[BM * LDS_S];
        __shared__ _Float16 Bs1[BN * LDS_S];
        for (int k0 = 0; k0 < K; k0 += BK) {
            #pragma unroll
            for (int i = 0; i < 16; ++i) {
                int idx = tid + i * 256;
                int r = idx >> 5, c = idx & 31;
                int gr = bm + r, gc = k0 + c;
                int cr = gr < M ? gr : M - 1;
                int cc = gc < K ? gc : K - 1;
                float v = A[(size_t)cr * K + cc];
                As1[r * LDS_S + c] = (_Float16)((gr < M && gc < K) ? v : 0.0f);
            }
            #pragma unroll
            for (int i = 0; i < 8; ++i) {
                int idx = tid + i * 256;
                int r = idx >> 6, c = idx & 63;     // r = k, c = n
                int gk = k0 + r, gn = bn + c;
                int ck = gk < K ? gk : K - 1;
                int cn = gn < N ? gn : N - 1;
                float v = B[(size_t)ck * N + cn];
                Bs1[c * LDS_S + r] = (_Float16)((gk < K && gn < N) ? v : 0.0f);
            }
            __syncthreads();
            Frag a0, a1, b0, b1;
            #pragma unroll
            for (int q = 0; q < 8; ++q) {
                int kk = ((q & 4) ? 16 : 0) + halfsel * 8 + (q & 3) * 2;
                a0.f[q] = *(const float*)&As1[(wm + l15) * LDS_S + kk];
                a1.f[q] = *(const float*)&As1[(wm + 16 + l15) * LDS_S + kk];
                b0.f[q] = *(const float*)&Bs1[(wn + l15) * LDS_S + kk];
                b1.f[q] = *(const float*)&Bs1[(wn + 16 + l15) * LDS_S + kk];
            }
            acc00 = __builtin_amdgcn_wmma_f32_16x16x32_f16(false, a0.v, false, b0.v, (short)0, acc00, false, false);
            acc01 = __builtin_amdgcn_wmma_f32_16x16x32_f16(false, a0.v, false, b1.v, (short)0, acc01, false, false);
            acc10 = __builtin_amdgcn_wmma_f32_16x16x32_f16(false, a1.v, false, b0.v, (short)0, acc10, false, false);
            acc11 = __builtin_amdgcn_wmma_f32_16x16x32_f16(false, a1.v, false, b1.v, (short)0, acc11, false, false);
            __syncthreads();
        }
    }

    // ---- store: C layout VGPR r -> m = r + 8*halfsel, n = lane&15 ----
    const int n0g = bn + wn + l15;
    const int n1g = n0g + 16;
    #pragma unroll
    for (int mi = 0; mi < 2; ++mi) {
        const int mBase = bm + wm + mi * 16 + 8 * halfsel;
        #pragma unroll
        for (int r = 0; r < 8; ++r) {
            int m = mBase + r;
            if (!FULL && m >= M) continue;
            float* Crow = C + (size_t)m * N;
            float v0 = mi ? acc10[r] : acc00[r];
            float v1 = mi ? acc11[r] : acc01[r];
            if (FULL || n0g < N) {
                float v = v0;
                if (biasFlag) v += bias[n0g];
                if (accumFlag) v += Crow[n0g];
                Crow[n0g] = v;
            }
            if (FULL || n1g < N) {
                float v = v1;
                if (biasFlag) v += bias[n1g];
                if (accumFlag) v += Crow[n1g];
                Crow[n1g] = v;
            }
        }
    }
}

// ---------------- small utility kernels ----------------
__global__ void fill_f32(float* p, float v, size_t n) {
    size_t i = (size_t)blockIdx.x * blockDim.x + threadIdx.x;
    if (i < n) p[i] = v;
}
__global__ void fill_u32(unsigned* p, unsigned v, size_t n) {
    size_t i = (size_t)blockIdx.x * blockDim.x + threadIdx.x;
    if (i < n) p[i] = v;
}
__global__ void transpose_f32(const float* __restrict__ in, float* __restrict__ out,
                              int R, int C) {   // in[R,C] -> out[C,R]
    int i = blockIdx.x * blockDim.x + threadIdx.x;
    if (i >= R * C) return;
    int r = i / C, c = i % C;
    out[(size_t)c * R + r] = in[i];
}
__global__ void vec_add_f32(const float* a, const float* b, float* o, int n) {
    int i = blockIdx.x * blockDim.x + threadIdx.x;
    if (i < n) o[i] = a[i] + b[i];
}

// ---------------- graph preprocessing ----------------
__global__ void build_edges(const int* __restrict__ eidx, const float* __restrict__ eattr,
                            int* srcA, int* dstA, float* eaAll, float* cnt, float* asum) {
    int e = blockIdx.x * blockDim.x + threadIdx.x;
    if (e >= N_EDGESC) return;
    int s = eidx[e], d = eidx[N_EDGESC + e];
    srcA[e] = s; dstA[e] = d;
    float a0 = eattr[e * 3], a1 = eattr[e * 3 + 1], a2 = eattr[e * 3 + 2];
    eaAll[e * 3] = a0; eaAll[e * 3 + 1] = a1; eaAll[e * 3 + 2] = a2;
    atomicAdd(&cnt[d], 1.0f);
    atomicAdd(&asum[d * 3 + 0], a0);
    atomicAdd(&asum[d * 3 + 1], a1);
    atomicAdd(&asum[d * 3 + 2], a2);
}
__global__ void build_self_loops(int* srcA, int* dstA, float* eaAll,
                                 const float* cnt, const float* asum) {
    int i = blockIdx.x * blockDim.x + threadIdx.x;
    if (i >= N_NODESC) return;
    srcA[N_EDGESC + i] = i; dstA[N_EDGESC + i] = i;
    float c = cnt[i]; c = c > 1.0f ? c : 1.0f;
    eaAll[(size_t)(N_EDGESC + i) * 3 + 0] = asum[i * 3 + 0] / c;
    eaAll[(size_t)(N_EDGESC + i) * 3 + 1] = asum[i * 3 + 1] / c;
    eaAll[(size_t)(N_EDGESC + i) * 3 + 2] = asum[i * 3 + 2] / c;
}

// monotone-encoded float atomic max
__device__ __forceinline__ unsigned encMax(float v) {
    unsigned b = __float_as_uint(v);
    return (b & 0x80000000u) ? ~b : (b | 0x80000000u);
}
__device__ __forceinline__ float decMax(unsigned e) {
    unsigned b = (e & 0x80000000u) ? (e & 0x7FFFFFFFu) : ~e;
    return __uint_as_float(b);
}

// ---------------- GATv2 edge passes ----------------
__global__ void gat_logits(const int* __restrict__ srcA, const int* __restrict__ dstA,
                           const float* __restrict__ eaAll,
                           const float* __restrict__ xl, const float* __restrict__ xr,
                           const float* __restrict__ We, const float* __restrict__ att,
                           float* __restrict__ logitb, unsigned* __restrict__ maxb,
                           int heads, int outc) {
    int idx = blockIdx.x * blockDim.x + threadIdx.x;
    if (idx >= EAC * heads) return;
    int e = idx / heads, h = idx % heads;
    int s = srcA[e], d = dstA[e];
    int HC = heads * outc;
    float ea0 = eaAll[e * 3], ea1 = eaAll[e * 3 + 1], ea2 = eaAll[e * 3 + 2];
    const float* xls = xl + (size_t)s * HC + h * outc;
    const float* xrd = xr + (size_t)d * HC + h * outc;
    const float* Weh = We + h * outc;       // rows stride HC
    const float* ath = att + h * outc;
    float logit = 0.0f;
    for (int c = 0; c < outc; ++c) {
        float ep = ea0 * Weh[c] + ea1 * Weh[HC + c] + ea2 * Weh[2 * HC + c];
        float m = xls[c] + xrd[c] + ep;
        m = m > 0.0f ? m : NEG_SLOPEC * m;
        logit += m * ath[c];
    }
    logitb[idx] = logit;
    atomicMax(&maxb[d * heads + h], encMax(logit));
}
__global__ void gat_expsum(const int* __restrict__ dstA, const unsigned* __restrict__ maxb,
                           float* __restrict__ logitb, float* __restrict__ denom, int heads) {
    int idx = blockIdx.x * blockDim.x + threadIdx.x;
    if (idx >= EAC * heads) return;
    int e = idx / heads, h = idx % heads;
    int d = dstA[e];
    float a = expf(logitb[idx] - decMax(maxb[d * heads + h]));
    logitb[idx] = a;
    atomicAdd(&denom[d * heads + h], a);
}
__global__ void gat_aggregate(const int* __restrict__ srcA, const int* __restrict__ dstA,
                              const float* __restrict__ xl, const float* __restrict__ logitb,
                              const float* __restrict__ denom, float* __restrict__ agg,
                              int heads, int outc) {
    size_t idx = (size_t)blockIdx.x * blockDim.x + threadIdx.x;
    int HC = heads * outc;
    if (idx >= (size_t)EAC * HC) return;
    int e = (int)(idx / HC), r = (int)(idx % HC);
    int h = r / outc;
    int s = srcA[e], d = dstA[e];
    float alpha = logitb[(size_t)e * heads + h] / (denom[d * heads + h] + 1e-16f);
    atomicAdd(&agg[(size_t)d * HC + r], xl[(size_t)s * HC + r] * alpha);
}
__global__ void bias_elu(const float* __restrict__ agg, const float* __restrict__ bias,
                         float* __restrict__ out, int HC, size_t total) {
    size_t i = (size_t)blockIdx.x * blockDim.x + threadIdx.x;
    if (i >= total) return;
    float v = agg[i] + bias[i % HC];
    out[i] = v > 0.0f ? v : (expf(v) - 1.0f);
}

// ---------------- LSTM cell pointwise (gates already in G) ----------------
__global__ void lstm_cell(const float* __restrict__ G, float* __restrict__ h,
                          float* __restrict__ c, float* __restrict__ outSeq) {
    int idx = blockIdx.x * blockDim.x + threadIdx.x;
    if (idx >= N_NODESC * LSTM_HC) return;
    int n = idx >> 8, j = idx & 255;
    const float* g = G + (size_t)n * LSTM_GC;
    float ig = 1.0f / (1.0f + expf(-g[j]));
    float fg = 1.0f / (1.0f + expf(-g[LSTM_HC + j]));
    float gg = tanhf(g[2 * LSTM_HC + j]);
    float og = 1.0f / (1.0f + expf(-g[3 * LSTM_HC + j]));
    float cn = fg * c[idx] + ig * gg;
    float hn = og * tanhf(cn);
    c[idx] = cn; h[idx] = hn;
    outSeq[(size_t)n * (2 * LSTM_HC) + j] = hn;   // caller offsets for direction
}

__global__ void gather_hcomb(const float* __restrict__ lastOut, float* __restrict__ hcomb) {
    int i = blockIdx.x * blockDim.x + threadIdx.x;
    if (i >= N_NODESC * 2 * LSTM_HC) return;
    int n = i / (2 * LSTM_HC), j = i % (2 * LSTM_HC);
    size_t srcIdx = (j < LSTM_HC)
        ? ((size_t)(SEQC - 1) * N_NODESC * 2 * LSTM_HC + (size_t)n * 2 * LSTM_HC + j)
        : ((size_t)n * 2 * LSTM_HC + j);
    hcomb[i] = lastOut[srcIdx];
}

// ---------------- host-side orchestration ----------------
static inline void launch_gemm(const float* A, const float* B, const float* bias, float* C,
                               int M, int N, int K, int accum, int biasFlag, hipStream_t s) {
    dim3 grid((N + BN - 1) / BN, (M + BM - 1) / BM);
    if ((M % BM == 0) && (N % BN == 0) && (K % BK == 0))
        gemm_wmma_f16<true><<<grid, 256, 0, s>>>(A, B, bias, C, M, N, K, accum, biasFlag);
    else
        gemm_wmma_f16<false><<<grid, 256, 0, s>>>(A, B, bias, C, M, N, K, accum, biasFlag);
}
static inline int blks(size_t n) { return (int)((n + 255) / 256); }

extern "C" void kernel_launch(void* const* d_in, const int* in_sizes, int n_in,
                              void* d_out, int out_size, void* d_ws, size_t ws_size,
                              hipStream_t stream) {
    (void)in_sizes; (void)n_in; (void)out_size; (void)ws_size;
    // ---- input map (top-level insertion order; params pytree-flattened, sorted keys) ----
    const float* x     = (const float*)d_in[0];   // [32,2048,128]
    const int*   eidx  = (const int*)  d_in[1];   // [2,32768]
    const float* eattr = (const float*)d_in[2];   // [32768,3]
    const float* fcW   = (const float*)d_in[3];   // [512,10]
    const float* fcB   = (const float*)d_in[4];   // [10]
    const float* g1We  = (const float*)d_in[5];   // [3,512]
    const float* g1Wl  = (const float*)d_in[6];   // [128,512]
    const float* g1Wr  = (const float*)d_in[7];
    const float* g1att = (const float*)d_in[8];   // [8,64]
    const float* g1bias= (const float*)d_in[9];   // [512]
    const float* g1bl  = (const float*)d_in[10];
    const float* g1br  = (const float*)d_in[11];
    const float* g2We  = (const float*)d_in[12];  // [3,64]
    const float* g2Wl  = (const float*)d_in[13];  // [512,64]
    const float* g2Wr  = (const float*)d_in[14];
    const float* g2att = (const float*)d_in[15];  // [1,64]
    const float* g2bias= (const float*)d_in[16];  // [64]
    const float* g2bl  = (const float*)d_in[17];
    const float* g2br  = (const float*)d_in[18];
    auto lstmW = [&](int layer, int isFwd, int which) -> const float* {
        // per layer: {'bwd','fwd'} sorted -> bwd first; tuple (Wih, Whh, bih, bhh)
        return (const float*)d_in[19 + layer * 8 + (isFwd ? 4 : 0) + which];
    };

    // ---- workspace carve ----
    float* w = (float*)d_ws;
    size_t off = 0;
    auto take = [&](size_t n) { float* p = w + off; off += n; return p; };
    int*   srcA  = (int*)take(EAC);
    int*   dstA  = (int*)take(EAC);
    float* eaAll = take((size_t)EAC * 3);
    float* cnt   = take(N_NODESC);
    float* asum  = take((size_t)N_NODESC * 3);
    float* xl    = take((size_t)N_NODESC * HC1C);
    float* xr    = take((size_t)N_NODESC * HC1C);
    float* agg   = take((size_t)N_NODESC * HC1C);
    float* h1    = take((size_t)N_NODESC * HC1C);
    float* logitb= take((size_t)EAC * HEADSC);
    unsigned* maxb = (unsigned*)take((size_t)N_NODESC * HEADSC);
    float* denom = take((size_t)N_NODESC * HEADSC);
    float* emb   = take((size_t)SEQC * N_NODESC * HIDC);
    float* seqA  = take((size_t)SEQC * N_NODESC * 2 * LSTM_HC);
    float* seqB  = take((size_t)SEQC * N_NODESC * 2 * LSTM_HC);
    float* G     = take((size_t)N_NODESC * LSTM_GC);
    float* hbuf  = take((size_t)N_NODESC * LSTM_HC);
    float* cbuf  = take((size_t)N_NODESC * LSTM_HC);
    float* hcomb = take((size_t)N_NODESC * 2 * LSTM_HC);
    float* wT_ih[N_LAYERSC][2];
    float* wT_hh[N_LAYERSC][2];
    float* bsum [N_LAYERSC][2];
    for (int l = 0; l < N_LAYERSC; ++l) {
        int Din = (l == 0) ? HIDC : 2 * LSTM_HC;
        for (int dir = 0; dir < 2; ++dir) {
            wT_ih[l][dir] = take((size_t)Din * LSTM_GC);
            wT_hh[l][dir] = take((size_t)LSTM_HC * LSTM_GC);
            bsum [l][dir] = take(LSTM_GC);
        }
    }

    // ---- graph preprocessing: self-loop attrs = mean of incoming ----
    fill_f32<<<blks(N_NODESC), 256, 0, stream>>>(cnt, 0.0f, N_NODESC);
    fill_f32<<<blks((size_t)N_NODESC * 3), 256, 0, stream>>>(asum, 0.0f, (size_t)N_NODESC * 3);
    build_edges<<<blks(N_EDGESC), 256, 0, stream>>>(eidx, eattr, srcA, dstA, eaAll, cnt, asum);
    build_self_loops<<<blks(N_NODESC), 256, 0, stream>>>(srcA, dstA, eaAll, cnt, asum);

    // ---- LSTM weight prep: transposes + bias sums ----
    for (int l = 0; l < N_LAYERSC; ++l) {
        int Din = (l == 0) ? HIDC : 2 * LSTM_HC;
        for (int dir = 0; dir < 2; ++dir) {
            int isFwd = (dir == 0);
            transpose_f32<<<blks((size_t)LSTM_GC * Din), 256, 0, stream>>>(
                lstmW(l, isFwd, 0), wT_ih[l][dir], LSTM_GC, Din);
            transpose_f32<<<blks((size_t)LSTM_GC * LSTM_HC), 256, 0, stream>>>(
                lstmW(l, isFwd, 1), wT_hh[l][dir], LSTM_GC, LSTM_HC);
            vec_add_f32<<<blks(LSTM_GC), 256, 0, stream>>>(
                lstmW(l, isFwd, 2), lstmW(l, isFwd, 3), bsum[l][dir], LSTM_GC);
        }
    }

    // ---- temporal GAT encoder ----
    for (int t = 0; t < SEQC; ++t) {
        const float* xt = x + (size_t)t * N_NODESC * IN_CHC;
        // GAT1 (heads=8, outc=64)
        launch_gemm(xt, g1Wl, g1bl, xl, N_NODESC, HC1C, IN_CHC, 0, 1, stream);
        launch_gemm(xt, g1Wr, g1br, xr, N_NODESC, HC1C, IN_CHC, 0, 1, stream);
        fill_u32<<<blks((size_t)N_NODESC * HEADSC), 256, 0, stream>>>(
            maxb, 0x007FFFFFu /* enc(-inf) */, (size_t)N_NODESC * HEADSC);
        fill_f32<<<blks((size_t)N_NODESC * HEADSC), 256, 0, stream>>>(
            denom, 0.0f, (size_t)N_NODESC * HEADSC);
        fill_f32<<<blks((size_t)N_NODESC * HC1C), 256, 0, stream>>>(
            agg, 0.0f, (size_t)N_NODESC * HC1C);
        gat_logits<<<blks((size_t)EAC * HEADSC), 256, 0, stream>>>(
            srcA, dstA, eaAll, xl, xr, g1We, g1att, logitb, maxb, HEADSC, HIDC);
        gat_expsum<<<blks((size_t)EAC * HEADSC), 256, 0, stream>>>(
            dstA, maxb, logitb, denom, HEADSC);
        gat_aggregate<<<blks((size_t)EAC * HC1C), 256, 0, stream>>>(
            srcA, dstA, xl, logitb, denom, agg, HEADSC, HIDC);
        bias_elu<<<blks((size_t)N_NODESC * HC1C), 256, 0, stream>>>(
            agg, g1bias, h1, HC1C, (size_t)N_NODESC * HC1C);
        // GAT2 (heads=1, outc=64) -> emb[t]
        launch_gemm(h1, g2Wl, g2bl, xl, N_NODESC, HIDC, HC1C, 0, 1, stream);
        launch_gemm(h1, g2Wr, g2br, xr, N_NODESC, HIDC, HC1C, 0, 1, stream);
        fill_u32<<<blks(N_NODESC), 256, 0, stream>>>(maxb, 0x007FFFFFu, N_NODESC);
        fill_f32<<<blks(N_NODESC), 256, 0, stream>>>(denom, 0.0f, N_NODESC);
        fill_f32<<<blks((size_t)N_NODESC * HIDC), 256, 0, stream>>>(
            agg, 0.0f, (size_t)N_NODESC * HIDC);
        gat_logits<<<blks(EAC), 256, 0, stream>>>(
            srcA, dstA, eaAll, xl, xr, g2We, g2att, logitb, maxb, 1, HIDC);
        gat_expsum<<<blks(EAC), 256, 0, stream>>>(dstA, maxb, logitb, denom, 1);
        gat_aggregate<<<blks((size_t)EAC * HIDC), 256, 0, stream>>>(
            srcA, dstA, xl, logitb, denom, agg, 1, HIDC);
        bias_elu<<<blks((size_t)N_NODESC * HIDC), 256, 0, stream>>>(
            agg, g2bias, emb + (size_t)t * N_NODESC * HIDC, HIDC, (size_t)N_NODESC * HIDC);
    }

    // ---- 4-layer bidirectional LSTM over T=32 ----
    const float* inSeq = emb;
    int Din = HIDC;
    float* outSeq = seqA;
    for (int l = 0; l < N_LAYERSC; ++l) {
        outSeq = (l & 1) ? seqB : seqA;
        for (int dir = 0; dir < 2; ++dir) {          // 0 = fwd, 1 = bwd
            int isFwd = (dir == 0);
            fill_f32<<<blks((size_t)N_NODESC * LSTM_HC), 256, 0, stream>>>(
                hbuf, 0.0f, (size_t)N_NODESC * LSTM_HC);
            fill_f32<<<blks((size_t)N_NODESC * LSTM_HC), 256, 0, stream>>>(
                cbuf, 0.0f, (size_t)N_NODESC * LSTM_HC);
            for (int s2 = 0; s2 < SEQC; ++s2) {
                int t = isFwd ? s2 : (SEQC - 1 - s2);
                // G = x_t @ Wih^T + (bih + bhh)
                launch_gemm(inSeq + (size_t)t * N_NODESC * Din, wT_ih[l][dir],
                            bsum[l][dir], G, N_NODESC, LSTM_GC, Din, 0, 1, stream);
                // G += h @ Whh^T
                launch_gemm(hbuf, wT_hh[l][dir], nullptr, G,
                            N_NODESC, LSTM_GC, LSTM_HC, 1, 0, stream);
                lstm_cell<<<blks((size_t)N_NODESC * LSTM_HC), 256, 0, stream>>>(
                    G, hbuf, cbuf,
                    outSeq + (size_t)t * N_NODESC * 2 * LSTM_HC + (isFwd ? 0 : LSTM_HC));
            }
        }
        inSeq = outSeq;
        Din = 2 * LSTM_HC;
    }

    // ---- head: concat(h_fwd@T-1, h_bwd@0) @ fcW + fcB ----
    gather_hcomb<<<blks((size_t)N_NODESC * 2 * LSTM_HC), 256, 0, stream>>>(outSeq, hcomb);
    launch_gemm(hcomb, fcW, fcB, (float*)d_out, N_NODESC, N_CLASSESC, 2 * LSTM_HC, 0, 1, stream);
}
